// MultiHeadAttention_38989713113233
// MI455X (gfx1250) — matrix operand order repaired and checked by reading, not compiled, not run on hardware
//
#include <hip/hip_runtime.h>
#include <stdint.h>

// ---------------------------------------------------------------------------
// Types for CDNA5 WMMA (wave32): V_WMMA_F32_16X16X32_BF16
// ---------------------------------------------------------------------------
typedef __attribute__((ext_vector_type(16))) __bf16   v16bf;
typedef __attribute__((ext_vector_type(8)))  float    v8f;
typedef __attribute__((ext_vector_type(8)))  uint16_t ush8;   // 16 bytes

union BF16Frag {
  v16bf    v;
  ush8     h[2];
  uint16_t u[16];
};

__device__ __forceinline__ uint16_t f2bf(float f) {
  uint32_t u = __float_as_uint(f);
  u += 0x7FFFu + ((u >> 16) & 1u);   // round-to-nearest-even
  return (uint16_t)(u >> 16);
}

__device__ __forceinline__ v8f wmma_bf16_f32(v16bf a, v16bf b, v8f c) {
  // (neg_a, A, neg_b, B, c_mod, C, reuse_a, reuse_b)
  return __builtin_amdgcn_wmma_f32_16x16x32_bf16(false, a, false, b, (short)0, c,
                                                 false, false);
}

// CDNA5 async global->LDS copy (ASYNCcnt-tracked, no VGPR data bounce).
// Per-lane: LDS[lds_dst] = MEM[gsrc], 16 bytes.
__device__ __forceinline__ void async_ld_b128(void* lds_dst, const void* gsrc) {
  uint32_t l = (uint32_t)(uintptr_t)lds_dst;   // generic LDS addr truncates to LDS offset
  asm volatile("global_load_async_to_lds_b128 %0, %1, off"
               :: "v"(l), "v"(gsrc)
               : "memory");
}

__device__ __forceinline__ void wait_async0() {
#if __has_builtin(__builtin_amdgcn_s_wait_asynccnt)
  __builtin_amdgcn_s_wait_asynccnt(0);
#else
  asm volatile("s_wait_asynccnt 0x0" ::: "memory");
#endif
}

// Problem constants
#define BSZ   2
#define TSEQ  2048
#define DMODEL 1024
#define NHEAD 16
#define HDIM  64
#define MROWS (BSZ * TSEQ)       // 4096
#define NCOL  1024               // N*H == D

// ---------------------------------------------------------------------------
// Kernel: f32 -> bf16 convert (identity layout)
// ---------------------------------------------------------------------------
__global__ __launch_bounds__(256) void k_cvt_bf16(const float* __restrict__ src,
                                                  uint16_t* __restrict__ dst, int n) {
  int i = blockIdx.x * 256 + threadIdx.x;
  if (i < n) dst[i] = f2bf(src[i]);
}

// w_projection [N, D, H] -> Wp[k = n*H+h][d]  (row-major [1024 x 1024] bf16)
__global__ __launch_bounds__(256) void k_pack_wproj(const float* __restrict__ src,
                                                    uint16_t* __restrict__ dst) {
  int i = blockIdx.x * 256 + threadIdx.x;   // 1M elements, exact grid
  int k = i >> 10, d = i & 1023;
  int n = k >> 6, h = k & 63;
  dst[i] = f2bf(src[((size_t)n * DMODEL + d) * HDIM + h]);
}

// ---------------------------------------------------------------------------
// WMMA GEMM: C[4096 x 1024] = A_bf16[4096 x 1024] * B_bf16[1024 x 1024] + bias
// Block tile 128x64, BK=32, 256 threads (8 waves); wave w -> rows w*16..w*16+15,
// all 64 cols (4 accumulators reusing one A fragment). A tile staged via
// GLOBAL_LOAD_ASYNC_TO_LDS_B128; B tile scattered into WMMA-B fragment order.
// layout==0: out_f32 row-major [M x 1024]
// layout==1: out_bnth bf16 scattered to [B, N, T, H]  (row=b*T+t, col=n*H+h)
// ---------------------------------------------------------------------------
__global__ __launch_bounds__(256)
void k_gemm_wmma(const uint16_t* __restrict__ A, const uint16_t* __restrict__ B,
                 const float* __restrict__ bias, float* __restrict__ out_f32,
                 uint16_t* __restrict__ out_bnth, int layout) {
  __shared__ __attribute__((aligned(16))) uint16_t As[128 * 40];   // rows padded to 40 halfs
  __shared__ __attribute__((aligned(16))) uint16_t Bs[4 * 32 * 16]; // fragment-order B tiles

  const int tid = threadIdx.x;
  const int w = tid >> 5;
  const int l = tid & 31;
  const int m0 = blockIdx.y * 128;
  const int n0 = blockIdx.x * 64;

  v8f acc[4];
#pragma unroll
  for (int c = 0; c < 4; ++c)
#pragma unroll
    for (int r = 0; r < 8; ++r) acc[c][r] = 0.0f;

  const int mA = l & 15;
  const int k0A = (l < 16) ? 0 : 16;

  for (int kb = 0; kb < DMODEL; kb += 32) {
    // ---- async A tile staging: 128x32 halfs, 2x 16B async copies per thread ----
    {
      int c = tid * 2;
#pragma unroll
      for (int i = 0; i < 2; ++i, ++c) {
        int row = c >> 2;
        int co = (c & 3) * 8;
        async_ld_b128(&As[row * 40 + co],
                      A + (size_t)(m0 + row) * DMODEL + kb + co);
      }
    }
    // ---- cooperative B tile load into WMMA-B fragment order (overlaps async) ----
    {
      int kk = tid >> 3;
      int co = (tid & 7) * 8;
      ush8 dv = *(const ush8*)(B + (size_t)(kb + kk) * NCOL + n0 + co);
      const uint16_t* dp = (const uint16_t*)&dv;
      int lane_add = kk & 16;
      int jj = kk & 15;
#pragma unroll
      for (int j = 0; j < 8; ++j) {
        int nl = co + j;
        Bs[(nl >> 4) * 512 + ((nl & 15) + lane_add) * 16 + jj] = dp[j];
      }
    }
    if (kb + 32 < DMODEL) {   // global_prefetch_b8 for next B tile
      __builtin_prefetch(B + (size_t)(kb + 32 + (tid >> 3)) * NCOL + n0 + (tid & 7) * 8, 0, 1);
    }
    wait_async0();
    __syncthreads();

    // ---- A fragment (contiguous k order, as in Bs): halves 0-7 -> K=k0..k0+7,
    //      halves 8-15 -> K=k0+8..k0+15, k0 = lane<16 ? 0 : 16 ----
    BF16Frag a;
    a.h[0] = *(const ush8*)&As[(w * 16 + mA) * 40 + k0A];
    a.h[1] = *(const ush8*)&As[(w * 16 + mA) * 40 + k0A + 8];

#pragma unroll
    for (int c = 0; c < 4; ++c) {
      BF16Frag bfr;
      bfr.h[0] = *(const ush8*)&Bs[c * 512 + l * 16];
      bfr.h[1] = *(const ush8*)&Bs[c * 512 + l * 16 + 8];
      acc[c] = wmma_bf16_f32(a.v, bfr.v, acc[c]);
    }
    __syncthreads();
  }

  // ---- epilogue: C layout lane holds rows m0w+{0..7|8..15}, col = l&15 ----
  const int m_base = m0 + w * 16 + ((l < 16) ? 0 : 8);
#pragma unroll
  for (int c = 0; c < 4; ++c) {
    int col = n0 + c * 16 + (l & 15);
    float bcol = bias[col];
#pragma unroll
    for (int r = 0; r < 8; ++r) {
      float val = acc[c][r] + bcol;
      int row = m_base + r;
      if (layout == 0) {
        out_f32[(size_t)row * NCOL + col] = val;
      } else {
        int b = row >> 11, t = row & 2047;       // T = 2048
        int n = col >> 6, h = col & 63;          // H = 64
        out_bnth[(((size_t)(b * NHEAD + n) * TSEQ + t) << 6) + h] = f2bf(val);
      }
    }
  }
}

// ---------------------------------------------------------------------------
// Flash-attention kernel: grid (B*N, T/128), 256 threads (8 waves).
// Wave owns 16 query rows; loop over 32-key blocks staged in LDS
// (K via GLOBAL_LOAD_ASYNC_TO_LDS_B128, V scattered to fragment order).
// Q,K,V bf16 [B,N,T,H]; writes attn bf16 [B,T,N,H].
// ---------------------------------------------------------------------------
__global__ __launch_bounds__(256)
void k_attention(const uint16_t* __restrict__ Q, const uint16_t* __restrict__ Kp,
                 const uint16_t* __restrict__ V, uint16_t* __restrict__ attn_out) {
  __shared__ __attribute__((aligned(16))) uint16_t Ks[32 * 72];      // 32 keys x 64 halfs (+pad)
  __shared__ __attribute__((aligned(16))) uint16_t Vs[4 * 32 * 16];  // B-fragment order
  __shared__ __attribute__((aligned(16))) float    Pws[8 * 16 * 40]; // per-wave P transpose

  const int tid = threadIdx.x;
  const int w = tid >> 5;
  const int l = tid & 31;
  const int bn = blockIdx.x;            // b*16 + n
  const int b = bn >> 4, n = bn & 15;
  const size_t headoff = (size_t)bn * TSEQ * HDIM;
  const uint16_t* Qh = Q + headoff;
  const uint16_t* Kh = Kp + headoff;
  const uint16_t* Vh = V + headoff;
  const int q0 = blockIdx.y * 128;
  const int qrow = q0 + w * 16;
  const int m = l & 15;

  // Q A-fragments for both K-steps over H (h = 32s + {k0..k0+15})
  BF16Frag aQ[2];
#pragma unroll
  for (int s = 0; s < 2; ++s) {
    int k0 = ((l < 16) ? 0 : 16) + 32 * s;
    aQ[s].h[0] = *(const ush8*)(Qh + (size_t)(qrow + m) * HDIM + k0);
    aQ[s].h[1] = *(const ush8*)(Qh + (size_t)(qrow + m) * HDIM + k0 + 8);
  }

  v8f o[4];
  float mrun[8], lrun[8];
#pragma unroll
  for (int c = 0; c < 4; ++c)
#pragma unroll
    for (int r = 0; r < 8; ++r) o[c][r] = 0.0f;
#pragma unroll
  for (int r = 0; r < 8; ++r) { mrun[r] = -3.0e38f; lrun[r] = 0.0f; }

  for (int kb = 0; kb < TSEQ; kb += 32) {
    // ---- staging: K async to LDS, V scattered to fragment order ----
    {
      int kk = tid >> 3;
      int co = (tid & 7) * 8;
      async_ld_b128(&Ks[kk * 72 + co], Kh + (size_t)(kb + kk) * HDIM + co);
      ush8 vv = *(const ush8*)(Vh + (size_t)(kb + kk) * HDIM + co);
      const uint16_t* vp = (const uint16_t*)&vv;
      int lane_add = kk & 16;
      int jj = kk & 15;
#pragma unroll
      for (int j = 0; j < 8; ++j) {
        int nl = co + j;
        Vs[(nl >> 4) * 512 + ((nl & 15) + lane_add) * 16 + jj] = vp[j];
      }
      if (kb + 32 < TSEQ) {
        __builtin_prefetch(Vh + (size_t)(kb + 32 + kk) * HDIM + co, 0, 1);
      }
    }
    wait_async0();
    __syncthreads();

    // ---- S = Q * K^T for two 16-key sub-tiles (K-dim = H, 2 WMMA steps) ----
    v8f s0, s1;
#pragma unroll
    for (int r = 0; r < 8; ++r) { s0[r] = 0.0f; s1[r] = 0.0f; }
#pragma unroll
    for (int s = 0; s < 2; ++s) {
      int h0 = ((l < 16) ? 0 : 16) + 32 * s;
      int nl = l & 15;
      BF16Frag bK0, bK1;
      bK0.h[0] = *(const ush8*)&Ks[nl * 72 + h0];
      bK0.h[1] = *(const ush8*)&Ks[nl * 72 + h0 + 8];
      bK1.h[0] = *(const ush8*)&Ks[(16 + nl) * 72 + h0];
      bK1.h[1] = *(const ush8*)&Ks[(16 + nl) * 72 + h0 + 8];
      s0 = wmma_bf16_f32(aQ[s].v, bK0.v, s0);
      s1 = wmma_bf16_f32(aQ[s].v, bK1.v, s1);
    }

    // ---- online softmax over the 32-key block (rows across 16-lane halves) ----
    const int mloc_base = (l < 16) ? 0 : 8;
#pragma unroll
    for (int r = 0; r < 8; ++r) {
      float a0 = s0[r] * 0.125f;          // 1/sqrt(64)
      float a1 = s1[r] * 0.125f;
      float sm = fmaxf(a0, a1);
      sm = fmaxf(sm, __shfl_xor(sm, 1, 32));
      sm = fmaxf(sm, __shfl_xor(sm, 2, 32));
      sm = fmaxf(sm, __shfl_xor(sm, 4, 32));
      sm = fmaxf(sm, __shfl_xor(sm, 8, 32));
      float mnew = fmaxf(mrun[r], sm);
      float p0 = __expf(a0 - mnew);
      float p1 = __expf(a1 - mnew);
      float rs = p0 + p1;
      rs += __shfl_xor(rs, 1, 32);
      rs += __shfl_xor(rs, 2, 32);
      rs += __shfl_xor(rs, 4, 32);
      rs += __shfl_xor(rs, 8, 32);
      float alpha = __expf(mrun[r] - mnew);
      lrun[r] = lrun[r] * alpha + rs;
      mrun[r] = mnew;
      o[0][r] *= alpha; o[1][r] *= alpha; o[2][r] *= alpha; o[3][r] *= alpha;
      Pws[w * 640 + (mloc_base + r) * 40 + (l & 15)] = p0;
      Pws[w * 640 + (mloc_base + r) * 40 + 16 + (l & 15)] = p1;
    }

    // ---- transpose P (C-layout -> A-layout) through per-wave LDS scratch ----
    BF16Frag aP;
    {
      int k0p = (l < 16) ? 0 : 16;
#pragma unroll
      for (int j = 0; j < 8; ++j) {
        aP.u[j]     = f2bf(Pws[w * 640 + m * 40 + k0p + j]);
        aP.u[j + 8] = f2bf(Pws[w * 640 + m * 40 + k0p + 8 + j]);
      }
    }

    // ---- O += P * V ----
#pragma unroll
    for (int c = 0; c < 4; ++c) {
      BF16Frag bV;
      bV.h[0] = *(const ush8*)&Vs[c * 512 + l * 16];
      bV.h[1] = *(const ush8*)&Vs[c * 512 + l * 16 + 8];
      o[c] = wmma_bf16_f32(aP.v, bV.v, o[c]);
    }
    __syncthreads();
  }

  // ---- normalize and store attn bf16 [B, T, N, H] ----
  const int row_base = qrow + ((l < 16) ? 0 : 8);
#pragma unroll
  for (int r = 0; r < 8; ++r) {
    float inv = 1.0f / lrun[r];
    int row = row_base + r;
#pragma unroll
    for (int c = 0; c < 4; ++c) {
      int h = c * 16 + (l & 15);
      attn_out[(((size_t)(b * TSEQ + row) * NHEAD) + n) * HDIM + h] = f2bf(o[c][r] * inv);
    }
  }
}

// ---------------------------------------------------------------------------
// Host launcher
// ---------------------------------------------------------------------------
extern "C" void kernel_launch(void* const* d_in, const int* in_sizes, int n_in,
                              void* d_out, int out_size, void* d_ws, size_t ws_size,
                              hipStream_t stream) {
  const float* q  = (const float*)d_in[0];
  const float* v  = (const float*)d_in[1];
  const float* k  = (const float*)d_in[2];
  const float* wq = (const float*)d_in[3];
  const float* bq = (const float*)d_in[4];
  const float* wv = (const float*)d_in[5];
  const float* bv = (const float*)d_in[6];
  const float* wk = (const float*)d_in[7];
  const float* bk = (const float*)d_in[8];
  const float* wp = (const float*)d_in[9];
  const float* bp = (const float*)d_in[10];

  const size_t SZ_X = (size_t)MROWS * DMODEL * 2;   // 8 MB  (bf16 activations)
  const size_t SZ_W = (size_t)DMODEL * NCOL * 2;    // 2 MB  (bf16 weights)
  uint8_t* ws = (uint8_t*)d_ws;
  uint16_t* qA    = (uint16_t*)(ws);
  uint16_t* vA    = (uint16_t*)(ws + 1 * SZ_X);
  uint16_t* kA    = (uint16_t*)(ws + 2 * SZ_X);
  uint16_t* wqB   = (uint16_t*)(ws + 3 * SZ_X);
  uint16_t* wvB   = (uint16_t*)(ws + 3 * SZ_X + 1 * SZ_W);
  uint16_t* wkB   = (uint16_t*)(ws + 3 * SZ_X + 2 * SZ_W);
  uint16_t* wpB   = (uint16_t*)(ws + 3 * SZ_X + 3 * SZ_W);
  uint16_t* Qb    = (uint16_t*)(ws + 3 * SZ_X + 4 * SZ_W);
  uint16_t* Kb    = (uint16_t*)(ws + 4 * SZ_X + 4 * SZ_W);
  uint16_t* Vb    = (uint16_t*)(ws + 5 * SZ_X + 4 * SZ_W);
  uint16_t* attnB = (uint16_t*)(ws + 6 * SZ_X + 4 * SZ_W);   // total 64 MB

  const int nX = MROWS * DMODEL;     // 4,194,304
  const int nW = DMODEL * NCOL;      // 1,048,576
  k_cvt_bf16<<<(nX + 255) / 256, 256, 0, stream>>>(q, qA, nX);
  k_cvt_bf16<<<(nX + 255) / 256, 256, 0, stream>>>(v, vA, nX);
  k_cvt_bf16<<<(nX + 255) / 256, 256, 0, stream>>>(k, kA, nX);
  k_cvt_bf16<<<(nW + 255) / 256, 256, 0, stream>>>(wq, wqB, nW);
  k_cvt_bf16<<<(nW + 255) / 256, 256, 0, stream>>>(wv, wvB, nW);
  k_cvt_bf16<<<(nW + 255) / 256, 256, 0, stream>>>(wk, wkB, nW);
  k_pack_wproj<<<nW / 256, 256, 0, stream>>>(wp, wpB);

  dim3 ggrid(NCOL / 64, MROWS / 128);   // (16, 32)
  k_gemm_wmma<<<ggrid, 256, 0, stream>>>(qA, wqB, bq, nullptr, Qb, 1);
  k_gemm_wmma<<<ggrid, 256, 0, stream>>>(kA, wkB, bk, nullptr, Kb, 1);
  k_gemm_wmma<<<ggrid, 256, 0, stream>>>(vA, wvB, bv, nullptr, Vb, 1);

  k_attention<<<dim3(BSZ * NHEAD, TSEQ / 128), 256, 0, stream>>>(Qb, Kb, Vb, attnB);

  k_gemm_wmma<<<ggrid, 256, 0, stream>>>(attnB, wpB, bp, (float*)d_out, nullptr, 0);
}
